// MLPAttention_68813966017170
// MI455X (gfx1250) — matrix operand order repaired and checked by reading, not compiled, not run on hardware
//
#include <hip/hip_runtime.h>
#include <hip/hip_bf16.h>

typedef __attribute__((ext_vector_type(16))) __bf16 v16bf;
typedef __attribute__((ext_vector_type(8)))  __bf16 v8bf;
typedef __attribute__((ext_vector_type(8)))  float  v8f;
typedef unsigned int u32x4 __attribute__((ext_vector_type(4)));
typedef int          i32x4 __attribute__((ext_vector_type(4)));
typedef int          i32x8 __attribute__((ext_vector_type(8)));

#define B_   32
#define S_   2048
#define QD_  1024
#define VD_  1024
#define H_   1024
#define NEGV (-1e9f)

#define STILE 64
#define APAD  8
#define ALD   (VD_ + APAD)   // 1032 bf16 per A row (2064 B, 16B aligned)
#define BPAD  16
#define BLD   (VD_ + BPAD)   // 1040 bf16 per B row (2080 B): 2048 B data + 2x16 B pad

#if defined(__has_builtin)
#if __has_builtin(__builtin_amdgcn_tensor_load_to_lds)
#define HAVE_TDM 1
#endif
#if __has_builtin(__builtin_amdgcn_s_wait_tensorcnt)
#define HAVE_WAIT_TENSOR 1
#endif
#if __has_builtin(__builtin_amdgcn_tanhf)
#define HAVE_TANH 1
#endif
#endif

#if __has_include(<hip/amd_detail/amd_gfx1250_TDM.h>)
#define TDM_6ARG 1   // amdgpu-toolchain: 6-arg tensor_load_to_lds
#endif

__device__ __forceinline__ float fast_tanh(float x) {
#ifdef HAVE_TANH
  return __builtin_amdgcn_tanhf(x);   // V_TANH_F32 trans op
#else
  return tanhf(x);
#endif
}

__device__ __forceinline__ void wait_tensor_zero() {
#ifdef HAVE_TDM
#ifdef HAVE_WAIT_TENSOR
  __builtin_amdgcn_s_wait_tensorcnt(0);
#else
  asm volatile("s_wait_tensorcnt 0" ::: "memory");
#endif
#endif
}

#ifdef HAVE_TDM
// TDM: load one 16 x 1024 bf16 panel (rows stride VD_ elements in global)
// into LDS at byte address lds_addr, with 16B LDS padding every 1024B so the
// effective LDS row stride is BLD*2 = 2080 bytes.
__device__ __forceinline__ void tdm_load_panel(const __bf16* gsrc,
                                               unsigned lds_addr) {
  unsigned long long ga = (unsigned long long)(uintptr_t)gsrc;
  u32x4 g0;
  g0[0] = 1u;                                   // count=1, load, no gather
  g0[1] = lds_addr;                             // D#.lds_addr
  g0[2] = (unsigned)(ga & 0xffffffffu);         // global_addr[31:0]
  g0[3] = (unsigned)((ga >> 32) & 0x1ffffffu)   // global_addr[56:32]
        | (2u << 30);                           // type = 2 (image)
  i32x8 g1;
  // wg_mask=0 | data_size=1(2B)<<16 | pad_en<<20 | pad_interval=7(256dw)<<22
  // | pad_amount=3(4dw=16B)<<25
  g1[0] = (1 << 16) | (1 << 20) | (7 << 22) | (3 << 25);
  g1[1] = (VD_ & 0xffff) << 16;                 // barrier_addr=0 | dim0.lo<<16
  g1[2] = ((VD_ >> 16) & 0xffff) | (16 << 16);  // dim0.hi | tensor_dim1=16
  g1[3] = (VD_ & 0xffff) << 16;                 // dim1.hi=0 | tile_dim0=1024
  g1[4] = 16;                                   // tile_dim1=16 | tile_dim2=0
  g1[5] = VD_;                                  // tensor_dim0_stride[31:0]
  g1[6] = 0;                                    // stride0.hi | stride1.lo
  g1[7] = 0;                                    // stride1.hi
  i32x4 z4 = {0, 0, 0, 0};
#ifdef TDM_6ARG
  i32x8 z8 = {0, 0, 0, 0, 0, 0, 0, 0};
  __builtin_amdgcn_tensor_load_to_lds(g0, g1, z4, z4, z8, 0);
#else
  __builtin_amdgcn_tensor_load_to_lds(g0, g1, z4, z4, 0);
#endif
}
#endif

// ---------------- hq = query @ Wq + bq  (tiny) ----------------
__global__ __launch_bounds__(256) void k_hq(const float* __restrict__ q,
                                            const float* __restrict__ Wq,
                                            const float* __restrict__ bq,
                                            float* __restrict__ hq) {
  int idx = blockIdx.x * 256 + threadIdx.x;   // over B_*H_
  int b = idx >> 10, h = idx & (H_ - 1);
  const float* qr = q + (size_t)b * QD_;
  float s = bq[h];
  for (int k = 0; k < QD_; ++k) s += qr[k] * Wq[(size_t)k * H_ + h];
  hq[idx] = s;
}

// ---------------- WvT[h][k] = bf16(Wv[k][h]) ----------------
__global__ __launch_bounds__(256) void k_wvt(const float* __restrict__ Wv,
                                             __bf16* __restrict__ WvT) {
  int idx = blockIdx.x * 256 + threadIdx.x;   // over H_*VD_
  int h = idx >> 10, k = idx & (VD_ - 1);
  WvT[(size_t)h * VD_ + k] = (__bf16)Wv[(size_t)k * H_ + h];
}

// ---------------- fused hv-GEMM + tanh-dot -> masked logits ----------------
__global__ __launch_bounds__(128) void k_main(const float* __restrict__ value,
                                              const int* __restrict__ mask,
                                              const float* __restrict__ hq,
                                              const __bf16* __restrict__ WvT,
                                              const float* __restrict__ w,
                                              float* __restrict__ logits) {
  extern __shared__ char smem[];
  __bf16* A   = (__bf16*)smem;                  // [STILE][ALD] bf16 value tile
  __bf16* Bp0 = A + (size_t)STILE * ALD;        // [16][BLD] panel buffer 0
  __bf16* Bp1 = Bp0 + (size_t)16 * BLD;         // [16][BLD] panel buffer 1
  float*  red = (float*)(Bp1 + (size_t)16 * BLD);  // [STILE] logit reduction

  const int tid = threadIdx.x;
  const int b   = blockIdx.y;
  const int s0  = blockIdx.x * STILE;
  const int wv  = tid >> 5;         // wave id: owns rows 16*wv .. 16*wv+15

  // Stage value tile (fp32 -> bf16), read once from HBM.
  const float* vsrc = value + ((size_t)b * S_ + s0) * VD_;
  for (int f = tid; f < STILE * VD_; f += 128) {
    int r = f >> 10, cc = f & (VD_ - 1);
    A[r * ALD + cc] = (__bf16)vsrc[(size_t)r * VD_ + cc];
  }
  if (tid < STILE) red[tid] = 0.f;

#ifdef HAVE_TDM
  const unsigned ldsB0 = (unsigned)(uintptr_t)Bp0;
  const unsigned ldsB1 = (unsigned)(uintptr_t)Bp1;
#endif

  // Stage B panel for h-tile `ht`: padded layout, element k of row r lives at
  // byte r*2080 + 2k + (k>=512 ? 16 : 0).
  auto stageB = [&](int ht, __bf16* dst, unsigned dstLds) {
#ifdef HAVE_TDM
    (void)dst;
    if (wv == 0) tdm_load_panel(WvT + (size_t)(ht * 16) * VD_, dstLds);
#else
    (void)dstLds;
    const int h0 = ht * 16;
    for (int f = tid; f < 16 * (VD_ / 8); f += 128) {
      int r = f >> 7, cc = (f & 127) * 8;
      *(v8bf*)(dst + (size_t)r * BLD + cc + (cc >= 512 ? 8 : 0)) =
          *(const v8bf*)(WvT + ((size_t)(h0 + r)) * VD_ + cc);
    }
#endif
  };

#ifdef HAVE_TDM
  stageB(0, Bp0, ldsB0);
#else
  stageB(0, Bp0, 0u);
#endif
  wait_tensor_zero();
  __syncthreads();   // value tile + panel 0 + red[] ready

  const int lane = tid & 31;
  const int hlf  = lane >> 4;       // lane half
  const int col  = lane & 15;       // N column / A row-in-block

  // A operand: lane holds row (16*wv+col); K = k0 + hlf*16 + e
  const __bf16* aptr = A + (size_t)(wv * 16 + col) * ALD + hlf * 16;
  const int boff = col * BLD + hlf * 16;   // B operand base within a panel

  float acc[8] = {0.f, 0.f, 0.f, 0.f, 0.f, 0.f, 0.f, 0.f};

  for (int ht = 0; ht < H_ / 16; ++ht) {
    const int h0 = ht * 16;
    const __bf16* cur = (ht & 1) ? Bp1 : Bp0;

    // Kick off DMA of the next panel into the other buffer.
    if (ht + 1 < H_ / 16) {
#ifdef HAVE_TDM
      stageB(ht + 1, (ht & 1) ? Bp0 : Bp1, (ht & 1) ? ldsB0 : ldsB1);
#else
      stageB(ht + 1, (ht & 1) ? Bp0 : Bp1, 0u);
#endif
    }

    const float hw = hq[(size_t)b * H_ + h0 + col];
    const float ww = w[h0 + col];
    const __bf16* bptr = cur + boff;

    v8f c = {};
#pragma unroll 4
    for (int k0 = 0; k0 < VD_; k0 += 32) {
      const __bf16* bp2 = bptr + k0 + (k0 >= 512 ? 8 : 0);  // LDS pad skip
      union { v16bf v; v8bf h2[2]; } av, bv;
      av.h2[0] = *(const v8bf*)(aptr + k0);
      av.h2[1] = *(const v8bf*)(aptr + k0 + 8);
      bv.h2[0] = *(const v8bf*)(bp2);
      bv.h2[1] = *(const v8bf*)(bp2 + 8);
      c = __builtin_amdgcn_wmma_f32_16x16x32_bf16(
              false, av.v, false, bv.v, (short)0, c, false, false);
    }
    // C element v: row = 16*wv + v + 8*hlf, column h0+col
#pragma unroll
    for (int v = 0; v < 8; ++v)
      acc[v] += ww * fast_tanh(hw + c[v]);

    wait_tensor_zero();  // next panel landed (issuing wave; others count==0)
    __syncthreads();     // all waves done with `cur`, next panel visible
  }

#pragma unroll
  for (int v = 0; v < 8; ++v)
    atomicAdd(&red[wv * 16 + hlf * 8 + v], acc[v]);
  __syncthreads();

  if (tid < STILE) {
    int s = s0 + tid;
    logits[(size_t)b * S_ + s] = mask[(size_t)b * S_ + s] ? red[tid] : NEGV;
  }
}

// ---------------- row softmax over S ----------------
__global__ __launch_bounds__(256) void k_softmax(const float* __restrict__ logits,
                                                 float* __restrict__ out) {
  __shared__ float swr[8];
  __shared__ float sbc;
  const int b = blockIdx.x, tid = threadIdx.x;
  const float* lrow = logits + (size_t)b * S_;
  float* orow = out + (size_t)b * S_;

  float m = -3.4e38f;
  for (int s = tid; s < S_; s += 256) m = fmaxf(m, lrow[s]);
  for (int o = 16; o > 0; o >>= 1) m = fmaxf(m, __shfl_xor(m, o, 32));
  if ((tid & 31) == 0) swr[tid >> 5] = m;
  __syncthreads();
  if (tid == 0) {
    float t = swr[0];
    for (int i = 1; i < 8; ++i) t = fmaxf(t, swr[i]);
    sbc = t;
  }
  __syncthreads();
  const float mx = sbc;

  float sum = 0.f;
  for (int s = tid; s < S_; s += 256) {
    float e = __expf(lrow[s] - mx);
    orow[s] = e;
    sum += e;
  }
  for (int o = 16; o > 0; o >>= 1) sum += __shfl_xor(sum, o, 32);
  __syncthreads();
  if ((tid & 31) == 0) swr[tid >> 5] = sum;
  __syncthreads();
  if (tid == 0) {
    float t = 0.f;
    for (int i = 0; i < 8; ++i) t += swr[i];
    sbc = t;
  }
  __syncthreads();
  const float inv = 1.f / sbc;
  for (int s = tid; s < S_; s += 256) orow[s] *= inv;
}

extern "C" void kernel_launch(void* const* d_in, const int* in_sizes, int n_in,
                              void* d_out, int out_size, void* d_ws, size_t ws_size,
                              hipStream_t stream) {
  const float* query = (const float*)d_in[0];
  const float* value = (const float*)d_in[1];
  const int*   mask  = (const int*)d_in[2];
  const float* Wq    = (const float*)d_in[3];
  const float* bq    = (const float*)d_in[4];
  const float* Wv    = (const float*)d_in[5];
  const float* w     = (const float*)d_in[6];
  float* out = (float*)d_out;

  char* ws = (char*)d_ws;
  float*  hq     = (float*)ws;                                   // 128 KB
  __bf16* WvT    = (__bf16*)(ws + 131072);                       // 2 MB
  float*  logits = (float*)(ws + 131072 + 2097152);              // 256 KB

  k_hq<<<dim3((B_ * H_) / 256), dim3(256), 0, stream>>>(query, Wq, bq, hq);
  k_wvt<<<dim3((H_ * VD_) / 256), dim3(256), 0, stream>>>(Wv, WvT);

  size_t lds = (size_t)STILE * ALD * 2 + 2 * (size_t)16 * BLD * 2 + STILE * 4;
  k_main<<<dim3(S_ / STILE, B_), dim3(128), lds, stream>>>(value, mask, hq, WvT,
                                                           w, logits);
  k_softmax<<<dim3(B_), dim3(256), 0, stream>>>(logits, out);
}
